// ClassificationGNN_72378788872351
// MI455X (gfx1250) — compile-verified, matrix operand order checked by reading, not run on hardware
//
#include <hip/hip_runtime.h>
#include <math.h>

// GCN 2-layer forward for MI455X (gfx1250, wave32, WMMA).
// N=100000, E=1600000, F_IN=F_HID=128, N_CLS=10.
// Roofline: ~4 GFLOP total, working set (~160MB) fits the 192MB L2 ->
// runtime is dominated by the edge gather/scatter (L2 atomics). GEMMs run
// on matrix cores (v_wmma_f32_16x16x32_f16, f32 accumulate) with weights
// pre-packed into the per-lane WMMA register layout so the inner loop is
// one 32B coalesced load per lane for B (no strided b32 gathers).

#define FIN  128
#define FHID 128
#define NCLS 10
#define H2STRIDE 16

typedef __attribute__((ext_vector_type(16))) _Float16 v16h;
typedef __attribute__((ext_vector_type(8)))  _Float16 v8h;
typedef __attribute__((ext_vector_type(8)))  float    v8f;

// ---------------- degree / norm ----------------

__global__ void k_deg_init(float* deg, int n) {
  int i = blockIdx.x * blockDim.x + threadIdx.x;
  if (i < n) deg[i] = 1.0f;  // self-loop contributes 1 to every node's degree
}

__global__ void k_deg_accum(const long long* __restrict__ dst, float* deg, int n_edges) {
  int e = blockIdx.x * blockDim.x + threadIdx.x;
  if (e < n_edges) atomicAdd(&deg[(int)dst[e]], 1.0f);
}

__global__ void k_deg_to_dinv(float* deg, int n) {
  int i = blockIdx.x * blockDim.x + threadIdx.x;
  if (i < n) {
    float d = deg[i];
    deg[i] = (d > 0.0f) ? rsqrtf(d) : 0.0f;
  }
}

// ---------------- weight pre-pack into WMMA B-operand lane layout --------
// For wave32 WMMA f16 (ISA 7.12.2): B tile 32x16, lane l holds column
// n = l&15 with K = khalf..khalf+7 and khalf+16..khalf+23, khalf = (l>>4)*8.
// Packed as [tileN][kChunk][lane][16 halves] -> 32B contiguous per lane.

__global__ void k_pack_w1(const float* __restrict__ W1, _Float16* __restrict__ W1p) {
  int t = blockIdx.x * blockDim.x + threadIdx.x;  // 8*4*32*16 = 16384
  if (t >= (FHID / 16) * (FIN / 32) * 32 * 16) return;
  int j    = t & 15;
  int lane = (t >> 4) & 31;
  int kc   = (t >> 9) & 3;
  int tile = t >> 11;
  int n = tile * 16 + (lane & 15);
  int k = kc * 32 + ((lane >> 4) << 3) + ((j < 8) ? j : (j + 8));
  W1p[t] = (_Float16)W1[(size_t)k * FHID + n];
}

__global__ void k_pack_w2(const float* __restrict__ W2, _Float16* __restrict__ W2p) {
  int t = blockIdx.x * blockDim.x + threadIdx.x;  // 1*4*32*16 = 2048
  if (t >= (FHID / 32) * 32 * 16) return;
  int j    = t & 15;
  int lane = (t >> 4) & 31;
  int kc   = t >> 9;
  int n = lane & 15;
  int k = kc * 32 + ((lane >> 4) << 3) + ((j < 8) ? j : (j + 8));
  W2p[t] = (n < NCLS) ? (_Float16)W2[(size_t)k * NCLS + n] : (_Float16)0.0f;
}

// ---------------- WMMA GEMM 1: h1 = x @ W1  (N x 128) ----------------
// One wave per 16x16 tile; 8 waves/block = one 16-row strip x 128 cols
// (A rows shared through cache). A: inline f32->f16 (v_cvt co-executes
// with XDL WMMA). B: single 32B vector load per lane from packed weights.

__global__ void k_gemm1(const float* __restrict__ x,
                        const _Float16* __restrict__ W1p,
                        float* __restrict__ h1, int n_nodes) {
  const int lane = threadIdx.x & 31;
  const int wave = threadIdx.x >> 5;      // 0..7 -> N tile
  const int row0 = blockIdx.x * 16;       // M tile
  const int n0   = wave * 16;
  if (row0 >= n_nodes) return;

  const int m     = lane & 15;
  const int khalf = (lane >> 4) << 3;     // +8 K offset for upper half-wave
  int arow = row0 + m;
  if (arow >= n_nodes) arow = n_nodes - 1;  // clamp (no tail: 100000%16==0)

  const v16h* __restrict__ bp = (const v16h*)W1p + (size_t)wave * 4 * 32 + lane;

  v8f acc = {};
  #pragma unroll
  for (int kc = 0; kc < FIN / 32; ++kc) {
    v16h a;
    const float* pa = x + (size_t)arow * FIN + kc * 32 + khalf;
    #pragma unroll
    for (int i = 0; i < 8; ++i) {
      a[i]     = (_Float16)pa[i];
      a[8 + i] = (_Float16)pa[16 + i];
    }
    v16h b = bp[kc * 32];
    acc = __builtin_amdgcn_wmma_f32_16x16x32_f16(
        false, a, false, b, (short)0, acc, false, false);
  }

  // C/D: VGPR j -> M = j (lanes 0-15) or 8+j (lanes 16-31), N = lane&15
  #pragma unroll
  for (int j = 0; j < 8; ++j) {
    int row = row0 + khalf + j;
    if (row < n_nodes) h1[(size_t)row * FHID + n0 + m] = acc[j];
  }
}

// ---------------- layer-1 aggregation ----------------

// agg1[i] = h1[i] * dinv[i]^2   (self-loop term; also initializes agg1)
__global__ void k_selfloop1(const float* __restrict__ h1,
                            const float* __restrict__ dinv,
                            float* __restrict__ agg1, int n_nodes) {
  long long t = (long long)blockIdx.x * blockDim.x + threadIdx.x;
  if (t >= (long long)n_nodes * 32) return;
  int i = (int)(t >> 5);
  int f = (int)(t & 31) * 4;
  float w = dinv[i] * dinv[i];
  float4 v = *(const float4*)(h1 + (size_t)i * FHID + f);
  float4 o = make_float4(v.x * w, v.y * w, v.z * w, v.w * w);
  *(float4*)(agg1 + (size_t)i * FHID + f) = o;
}

// one wave per edge, each lane owns 4 contiguous features (float4 gather,
// 4 global_atomic_add_f32 scatter) -> coalesced rows, L2-resident
__global__ void k_scatter1(const long long* __restrict__ src,
                           const long long* __restrict__ dst,
                           const float* __restrict__ dinv,
                           const float* __restrict__ h1,
                           float* __restrict__ agg1, int n_edges) {
  int lane = threadIdx.x & 31;
  long long e = (long long)blockIdx.x * 8 + (threadIdx.x >> 5);
  if (e >= n_edges) return;
  int s = (int)src[e], d = (int)dst[e];
  float w = dinv[s] * dinv[d];
  float4 v = *(const float4*)(h1 + (size_t)s * FHID + lane * 4);
  float* out = agg1 + (size_t)d * FHID + lane * 4;
  atomicAdd(out + 0, v.x * w);
  atomicAdd(out + 1, v.y * w);
  atomicAdd(out + 2, v.z * w);
  atomicAdd(out + 3, v.w * w);
}

// act(f16) = leaky_relu(agg1 + b1, 0.2) -- emits GEMM2's A operand directly
__global__ void k_bias_lrelu(const float* __restrict__ agg1,
                             const float* __restrict__ b1,
                             _Float16* __restrict__ acth, long long total) {
  long long t = (long long)blockIdx.x * blockDim.x + threadIdx.x;
  if (t >= total) return;
  float v = agg1[t] + b1[(int)(t & (FHID - 1))];
  acth[t] = (_Float16)((v >= 0.0f) ? v : 0.2f * v);
}

// ---------------- WMMA GEMM 2: h2 = act @ W2 (N x 10, stride-16 store) ----

__global__ void k_gemm2(const _Float16* __restrict__ acth,
                        const _Float16* __restrict__ W2p,
                        float* __restrict__ h2, int n_nodes) {
  const int lane = threadIdx.x & 31;
  const int wave = threadIdx.x >> 5;
  const int row0 = (blockIdx.x * 8 + wave) * 16;
  if (row0 >= n_nodes) return;

  const int m     = lane & 15;
  const int khalf = (lane >> 4) << 3;
  int arow = row0 + m;
  if (arow >= n_nodes) arow = n_nodes - 1;

  const v16h* __restrict__ bp = (const v16h*)W2p + lane;

  v8f acc = {};
  #pragma unroll
  for (int kc = 0; kc < FHID / 32; ++kc) {
    const _Float16* pa = acth + (size_t)arow * FHID + kc * 32 + khalf;
    v8h lo = *(const v8h*)pa;
    v8h hi = *(const v8h*)(pa + 16);
    v16h a = __builtin_shufflevector(lo, hi, 0, 1, 2, 3, 4, 5, 6, 7,
                                     8, 9, 10, 11, 12, 13, 14, 15);
    v16h b = bp[kc * 32];
    acc = __builtin_amdgcn_wmma_f32_16x16x32_f16(
        false, a, false, b, (short)0, acc, false, false);
  }

  #pragma unroll
  for (int j = 0; j < 8; ++j) {
    int row = row0 + khalf + j;
    if (row < n_nodes && m < NCLS) h2[(size_t)row * H2STRIDE + m] = acc[j];
  }
}

// ---------------- layer-2 aggregation + log-softmax ----------------

__global__ void k_selfloop2(const float* __restrict__ h2,
                            const float* __restrict__ dinv,
                            float* __restrict__ agg2, int n_nodes) {
  int i = blockIdx.x * blockDim.x + threadIdx.x;
  if (i >= n_nodes) return;
  float w = dinv[i] * dinv[i];
  #pragma unroll
  for (int c = 0; c < NCLS; ++c)
    agg2[(size_t)i * H2STRIDE + c] = h2[(size_t)i * H2STRIDE + c] * w;
}

__global__ void k_scatter2(const long long* __restrict__ src,
                           const long long* __restrict__ dst,
                           const float* __restrict__ dinv,
                           const float* __restrict__ h2,
                           float* __restrict__ agg2, int n_edges) {
  int e = blockIdx.x * blockDim.x + threadIdx.x;
  if (e >= n_edges) return;
  int s = (int)src[e], d = (int)dst[e];
  float w = dinv[s] * dinv[d];
  const float* hs = h2 + (size_t)s * H2STRIDE;
  float* out = agg2 + (size_t)d * H2STRIDE;
  #pragma unroll
  for (int c = 0; c < NCLS; ++c) atomicAdd(out + c, hs[c] * w);
}

__global__ void k_logsoftmax(const float* __restrict__ agg2,
                             const float* __restrict__ b2,
                             float* __restrict__ out, int n_nodes) {
  int i = blockIdx.x * blockDim.x + threadIdx.x;
  if (i >= n_nodes) return;
  float l[NCLS];
  float mx = -3.402823466e+38f;
  #pragma unroll
  for (int c = 0; c < NCLS; ++c) {
    l[c] = agg2[(size_t)i * H2STRIDE + c] + b2[c];
    mx = fmaxf(mx, l[c]);
  }
  float s = 0.0f;
  #pragma unroll
  for (int c = 0; c < NCLS; ++c) s += __expf(l[c] - mx);
  float ls = __logf(s);
  #pragma unroll
  for (int c = 0; c < NCLS; ++c) out[(size_t)i * NCLS + c] = l[c] - mx - ls;
}

// ---------------- launch ----------------

extern "C" void kernel_launch(void* const* d_in, const int* in_sizes, int n_in,
                              void* d_out, int out_size, void* d_ws, size_t ws_size,
                              hipStream_t stream) {
  const float*     x  = (const float*)d_in[0];
  const long long* ei = (const long long*)d_in[1];  // int64, shape (2, E)
  const float*     W1 = (const float*)d_in[2];
  const float*     b1 = (const float*)d_in[3];
  const float*     W2 = (const float*)d_in[4];
  const float*     b2 = (const float*)d_in[5];
  float*           out = (float*)d_out;

  const int n_nodes = in_sizes[0] / FIN;   // 100000
  const int n_edges = in_sizes[1] / 2;     // 1600000
  const long long* src = ei;
  const long long* dst = ei + n_edges;

  // workspace layout (~142 MB): W1p | W2p | dinv | h1 | agg1 | acth | h2 | agg2
  char* ws = (char*)d_ws;
  _Float16* W1p = (_Float16*)ws;                 ws += 16384 * sizeof(_Float16);
  _Float16* W2p = (_Float16*)ws;                 ws += 2048 * sizeof(_Float16);
  float* dinv = (float*)ws;                      ws += (((size_t)n_nodes + 63) & ~(size_t)63) * 4;
  float* h1   = (float*)ws;                      ws += (size_t)n_nodes * FHID * 4;
  float* agg1 = (float*)ws;                      ws += (size_t)n_nodes * FHID * 4;
  _Float16* acth = (_Float16*)ws;                ws += (size_t)n_nodes * FHID * 2;
  float* h2   = (float*)ws;                      ws += (size_t)n_nodes * H2STRIDE * 4;
  float* agg2 = (float*)ws;

  const int B = 256;

  // one-time weight pre-pack (runs concurrently-safe, tiny)
  k_pack_w1<<<16384 / B, B, 0, stream>>>(W1, W1p);
  k_pack_w2<<<2048 / B, B, 0, stream>>>(W2, W2p);

  // degree -> dinv (reuses dinv buffer as deg)
  k_deg_init<<<(n_nodes + B - 1) / B, B, 0, stream>>>(dinv, n_nodes);
  k_deg_accum<<<(n_edges + B - 1) / B, B, 0, stream>>>(dst, dinv, n_edges);
  k_deg_to_dinv<<<(n_nodes + B - 1) / B, B, 0, stream>>>(dinv, n_nodes);

  // layer 1
  int tilesM = (n_nodes + 15) / 16;                        // 6250
  k_gemm1<<<tilesM, B, 0, stream>>>(x, W1p, h1, n_nodes);  // 8 waves = 8 N-tiles
  {
    long long t = (long long)n_nodes * 32;
    k_selfloop1<<<(unsigned)((t + B - 1) / B), B, 0, stream>>>(h1, dinv, agg1, n_nodes);
  }
  k_scatter1<<<(n_edges + 7) / 8, B, 0, stream>>>(src, dst, dinv, h1, agg1, n_edges);
  {
    long long t = (long long)n_nodes * FHID;
    k_bias_lrelu<<<(unsigned)((t + B - 1) / B), B, 0, stream>>>(agg1, b1, acth, t);
  }

  // layer 2
  int tiles2 = (tilesM + 7) / 8;                           // 782 blocks, 8 tiles each
  k_gemm2<<<tiles2, B, 0, stream>>>(acth, W2p, h2, n_nodes);
  k_selfloop2<<<(n_nodes + B - 1) / B, B, 0, stream>>>(h2, dinv, agg2, n_nodes);
  k_scatter2<<<(n_edges + B - 1) / B, B, 0, stream>>>(src, dst, dinv, h2, agg2, n_edges);

  k_logsoftmax<<<(n_nodes + B - 1) / B, B, 0, stream>>>(agg2, b2, out, n_nodes);
}
